// TwinRGCNConv_34548716929228
// MI455X (gfx1250) — matrix operand order repaired and checked by reading, not compiled, not run on hardware
//
#include <hip/hip_runtime.h>

#define D 128
#define WAVES 8

typedef __attribute__((ext_vector_type(16))) __bf16 v16bf;
typedef __attribute__((ext_vector_type(8)))  float  v8f;

union BfTile { uint4 q[2]; unsigned int u[8]; v16bf v; };

__device__ __forceinline__ unsigned int pack2bf(float lo, float hi) {
    unsigned int a = __float_as_uint(lo), b = __float_as_uint(hi);
    a = (a + 0x7FFFu + ((a >> 16) & 1u)) >> 16;   // round-to-nearest-even bf16
    b = (b + 0x7FFFu + ((b >> 16) & 1u)) >> 16;
    return (a & 0xFFFFu) | (b << 16);
}

// ---------------------------------------------------------------------------
// Pack X[N,128] f32 into WMMA A-tile layout, optionally scaling rows by
// 1/max(cnt[row],1) (fused mean-divide for the aggregation buffers).
// Packed layout (dwords): out[((strip*4 + ktile)*32 + lane)*8 + j]
// A-tile (16-bit 16x32, this code's contiguous k order): lane l -> row (l&15),
//   K = kofs..kofs+15 (v0..7), kofs = 16*(l>>4).
// ---------------------------------------------------------------------------
__global__ void pack_x_kernel(const float* __restrict__ X,
                              const float* __restrict__ cnt,
                              unsigned int* __restrict__ out, int tiles /* strips*4 */) {
    const long i = (long)blockIdx.x * blockDim.x + threadIdx.x;   // tile*32 + lane
    if (i >= (long)tiles * 32) return;
    const int lane  = (int)(i & 31);
    const long tile = i >> 5;
    const int  t    = (int)(tile & 3);
    const long s    = tile >> 2;

    const long row  = s * 16 + (lane & 15);
    const int  kofs = t * 32 + 16 * (lane >> 4);

    const float* p = X + row * D + kofs;
    float4 a0 = *(const float4*)(p);
    float4 a1 = *(const float4*)(p + 4);
    float4 a2 = *(const float4*)(p + 8);
    float4 a3 = *(const float4*)(p + 12);

    if (cnt) {
        const float sc = 1.0f / fmaxf(cnt[row], 1.0f);
        a0.x *= sc; a0.y *= sc; a0.z *= sc; a0.w *= sc;
        a1.x *= sc; a1.y *= sc; a1.z *= sc; a1.w *= sc;
        a2.x *= sc; a2.y *= sc; a2.z *= sc; a2.w *= sc;
        a3.x *= sc; a3.y *= sc; a3.z *= sc; a3.w *= sc;
    }

    BfTile r;
    r.u[0] = pack2bf(a0.x, a0.y); r.u[1] = pack2bf(a0.z, a0.w);
    r.u[2] = pack2bf(a1.x, a1.y); r.u[3] = pack2bf(a1.z, a1.w);
    r.u[4] = pack2bf(a2.x, a2.y); r.u[5] = pack2bf(a2.z, a2.w);
    r.u[6] = pack2bf(a3.x, a3.y); r.u[7] = pack2bf(a3.z, a3.w);

    uint4* o = (uint4*)(out + i * 8);
    o[0] = r.q[0];
    o[1] = r.q[1];
}

// ---------------------------------------------------------------------------
// Pack W (optionally Wa+Wb) into WMMA B-tile layout.
// B[k,n] = W[n,k]; B-tile 32x16: lane l -> col n = (l&15), K = 16*(l>>4)+[0..15],
// VGPR j packs K pair (2j, 2j+1).  out[((nt*4 + ktile)*32 + lane)*8 + j]
// 1024 threads total per weight.
// ---------------------------------------------------------------------------
__global__ void pack_w_kernel(const float* __restrict__ Wa,
                              const float* __restrict__ Wb,
                              unsigned int* __restrict__ out) {
    const int i = blockIdx.x * blockDim.x + threadIdx.x;   // (nt*4 + t)*32 + lane
    if (i >= 8 * 4 * 32) return;
    const int lane = i & 31;
    const int t    = (i >> 5) & 3;
    const int nt   = i >> 7;

    const int n  = nt * 16 + (lane & 15);
    const int kb = t * 32 + 16 * (lane >> 4);

    const float* p = Wa + (long)n * D + kb;
    float4 b0 = *(const float4*)(p);
    float4 b1 = *(const float4*)(p + 4);
    float4 b2 = *(const float4*)(p + 8);
    float4 b3 = *(const float4*)(p + 12);

    if (Wb) {
        const float* q = Wb + (long)n * D + kb;
        float4 c0 = *(const float4*)(q);
        float4 c1 = *(const float4*)(q + 4);
        float4 c2 = *(const float4*)(q + 8);
        float4 c3 = *(const float4*)(q + 12);
        b0.x += c0.x; b0.y += c0.y; b0.z += c0.z; b0.w += c0.w;
        b1.x += c1.x; b1.y += c1.y; b1.z += c1.z; b1.w += c1.w;
        b2.x += c2.x; b2.y += c2.y; b2.z += c2.z; b2.w += c2.w;
        b3.x += c3.x; b3.y += c3.y; b3.z += c3.z; b3.w += c3.w;
    }

    BfTile r;
    r.u[0] = pack2bf(b0.x, b0.y); r.u[1] = pack2bf(b0.z, b0.w);
    r.u[2] = pack2bf(b1.x, b1.y); r.u[3] = pack2bf(b1.z, b1.w);
    r.u[4] = pack2bf(b2.x, b2.y); r.u[5] = pack2bf(b2.z, b2.w);
    r.u[6] = pack2bf(b3.x, b3.y); r.u[7] = pack2bf(b3.z, b3.w);

    uint4* o = (uint4*)(out + (long)i * 8);
    o[0] = r.q[0];
    o[1] = r.q[1];
}

// ---------------------------------------------------------------------------
// out[M,128] = unpack(pA1) @ unpack(pW1)^T (+ pA2 @ pW2^T) + bias
// Pure b128 loads + v_wmma: conversion already done by pack kernels.
// ---------------------------------------------------------------------------
__global__ __launch_bounds__(32 * WAVES)
void gemm_wmma_kernel(const unsigned int* __restrict__ pA1,
                      const unsigned int* __restrict__ pW1,
                      const unsigned int* __restrict__ pA2,
                      const unsigned int* __restrict__ pW2,
                      const float* __restrict__ bias,
                      float* __restrict__ out, int strips) {
    const int lane  = threadIdx.x & 31;
    const int wave  = threadIdx.x >> 5;
    const int strip = blockIdx.x * WAVES + wave;
    if (strip >= strips) return;               // uniform per wave: EXEC stays all-1s

    const int lm = lane & 15;
    const int lh = lane >> 4;

    // A base (uint4 units): ((strip*4 + t)*32 + lane)*2  -> per-t stride 64
    const long abase = ((long)strip * 4 * 32 + lane) * 2;
    const uint4* A1p = (const uint4*)pA1 + abase;

    BfTile A1[4], A2[4];
#pragma unroll
    for (int t = 0; t < 4; ++t) {
        A1[t].q[0] = A1p[t * 64];
        A1[t].q[1] = A1p[t * 64 + 1];
    }
    if (pA2) {
        const uint4* A2p = (const uint4*)pA2 + abase;
#pragma unroll
        for (int t = 0; t < 4; ++t) {
            A2[t].q[0] = A2p[t * 64];
            A2[t].q[1] = A2p[t * 64 + 1];
        }
    }

#pragma unroll
    for (int nt = 0; nt < 8; ++nt) {
        const int col = nt * 16 + lm;
        const float bv = bias[col];
        v8f acc = {bv, bv, bv, bv, bv, bv, bv, bv};

        const long wbase = ((long)nt * 4 * 32 + lane) * 2;  // uint4 units, per-t stride 64
        const uint4* B1p = (const uint4*)pW1 + wbase;
#pragma unroll
        for (int t = 0; t < 4; ++t) {
            BfTile B;
            B.q[0] = B1p[t * 64];
            B.q[1] = B1p[t * 64 + 1];
            acc = __builtin_amdgcn_wmma_f32_16x16x32_bf16(
                false, A1[t].v, false, B.v, (short)0, acc, false, false);
        }
        if (pA2) {
            const uint4* B2p = (const uint4*)pW2 + wbase;
#pragma unroll
            for (int t = 0; t < 4; ++t) {
                BfTile B;
                B.q[0] = B2p[t * 64];
                B.q[1] = B2p[t * 64 + 1];
                acc = __builtin_amdgcn_wmma_f32_16x16x32_bf16(
                    false, A2[t].v, false, B.v, (short)0, acc, false, false);
            }
        }

        // C/D layout: VGPR r -> row strip*16 + lh*8 + r, col nt*16 + lm
        const long rbase = (long)strip * 16 + lh * 8;
#pragma unroll
        for (int r = 0; r < 8; ++r)
            out[(rbase + r) * D + col] = acc[r];
    }
}

// One thread per (edge, 4-channel group): gather src chunk, HW fp32 atomics into dst.
__global__ void scatter_edges_kernel(const float* __restrict__ x,
                                     const int* __restrict__ src,
                                     const int* __restrict__ dst,
                                     float* __restrict__ agg,
                                     float* __restrict__ cnt, int E) {
    const long t = (long)blockIdx.x * blockDim.x + threadIdx.x;
    const long e = t >> 5;
    if (e >= E) return;
    const int g = (int)(t & 31) * 4;
    const int s = src[e], d = dst[e];
    const float4 v = *(const float4*)(x + (long)s * D + g);
    float* o = agg + (long)d * D + g;
    unsafeAtomicAdd(o + 0, v.x);
    unsafeAtomicAdd(o + 1, v.y);
    unsafeAtomicAdd(o + 2, v.z);
    unsafeAtomicAdd(o + 3, v.w);
    if (g == 0) unsafeAtomicAdd(cnt + d, 1.0f);
}

extern "C" void kernel_launch(void* const* d_in, const int* in_sizes, int n_in,
                              void* d_out, int out_size, void* d_ws, size_t ws_size,
                              hipStream_t stream) {
    const float* x_a      = (const float*)d_in[0];
    const float* x_b      = (const float*)d_in[1];
    const float* xp_a     = (const float*)d_in[2];
    const float* xp_b     = (const float*)d_in[3];
    const int*   src_ab   = (const int*)d_in[4];
    const int*   dst_ab   = (const int*)d_in[5];
    const int*   src_ba   = (const int*)d_in[6];
    const int*   dst_ba   = (const int*)d_in[7];
    const float* W_root_a = (const float*)d_in[8];
    const float* b_root_a = (const float*)d_in[9];
    const float* W_root_b = (const float*)d_in[10];
    const float* b_root_b = (const float*)d_in[11];
    const float* W_rel_ab = (const float*)d_in[12];
    const float* W_rel_ba = (const float*)d_in[13];

    const int ND = in_sizes[0];          // N * D
    const int Nn = ND / D;               // N
    const int E1 = in_sizes[4];
    const int E2 = in_sizes[6];
    const int strips = Nn / 16;          // 50000/16 = 3125
    const int tiles  = strips * 4;       // A k-tiles per matrix

    // ---- workspace layout ----
    float* aggAB = (float*)d_ws;                 // [N, D] f32
    float* aggBA = aggAB + ND;                   // [N, D] f32
    float* cntAB = aggBA + ND;                   // [N]
    float* cntBA = cntAB + Nn;                   // [N]
    unsigned int* pk = (unsigned int*)(cntBA + Nn);
    const long XP = (long)ND / 2;                // packed activation size (dwords)
    unsigned int* p_xa   = pk;            unsigned int* p_xb   = p_xa   + XP;
    unsigned int* p_xpa  = p_xb  + XP;    unsigned int* p_xpb  = p_xpa  + XP;
    unsigned int* p_agAB = p_xpb + XP;    unsigned int* p_agBA = p_agAB + XP;
    const int WP = D * D / 2;                    // packed weight size (dwords)
    unsigned int* w_ra   = p_agBA + XP;          // W_root_a
    unsigned int* w_rb   = w_ra + WP;            // W_root_b
    unsigned int* w_ab   = w_rb + WP;            // W_rel_ab
    unsigned int* w_ba   = w_ab + WP;            // W_rel_ba
    unsigned int* w_ca   = w_ba + WP;            // W_root_a + W_rel_ba
    unsigned int* w_cb   = w_ca + WP;            // W_root_b + W_rel_ab

    hipMemsetAsync(aggAB, 0, sizeof(float) * ((size_t)2 * ND + 2 * Nn), stream);

    // weights -> packed B layout (tiny: 1024 threads each)
    pack_w_kernel<<<4, 256, 0, stream>>>(W_root_a, nullptr,  w_ra);
    pack_w_kernel<<<4, 256, 0, stream>>>(W_root_b, nullptr,  w_rb);
    pack_w_kernel<<<4, 256, 0, stream>>>(W_rel_ab, nullptr,  w_ab);
    pack_w_kernel<<<4, 256, 0, stream>>>(W_rel_ba, nullptr,  w_ba);
    pack_w_kernel<<<4, 256, 0, stream>>>(W_root_a, W_rel_ba, w_ca);
    pack_w_kernel<<<4, 256, 0, stream>>>(W_root_b, W_rel_ab, w_cb);

    // activations -> packed A layout
    const long pth = (long)tiles * 32;
    const int  pblk = (int)((pth + 255) / 256);
    pack_x_kernel<<<pblk, 256, 0, stream>>>(x_a,  nullptr, p_xa,  tiles);
    pack_x_kernel<<<pblk, 256, 0, stream>>>(x_b,  nullptr, p_xb,  tiles);
    pack_x_kernel<<<pblk, 256, 0, stream>>>(xp_a, nullptr, p_xpa, tiles);
    pack_x_kernel<<<pblk, 256, 0, stream>>>(xp_b, nullptr, p_xpb, tiles);

    // sparse mean-aggregation (HW fp32 atomics)
    const long t1 = (long)E1 * 32, t2 = (long)E2 * 32;
    scatter_edges_kernel<<<(int)((t1 + 255) / 256), 256, 0, stream>>>(x_a, src_ab, dst_ab, aggAB, cntAB, E1);
    scatter_edges_kernel<<<(int)((t2 + 255) / 256), 256, 0, stream>>>(x_b, src_ba, dst_ba, aggBA, cntBA, E2);

    // fused mean-divide + pack
    pack_x_kernel<<<pblk, 256, 0, stream>>>(aggAB, cntAB, p_agAB, tiles);
    pack_x_kernel<<<pblk, 256, 0, stream>>>(aggBA, cntBA, p_agBA, tiles);

    const int blocks = (strips + WAVES - 1) / WAVES;
    float* out_a  = (float*)d_out;
    float* out_b  = out_a + ND;
    float* outp_a = out_b + ND;
    float* outp_b = outp_a + ND;

    // out_a  = x_a@W_root_a^T + agg_ba@W_rel_ba^T + b_root_a
    gemm_wmma_kernel<<<blocks, 32 * WAVES, 0, stream>>>(p_xa,  w_ra, p_agBA, w_ba, b_root_a, out_a,  strips);
    // out_b  = x_b@W_root_b^T + agg_ab@W_rel_ab^T + b_root_b
    gemm_wmma_kernel<<<blocks, 32 * WAVES, 0, stream>>>(p_xb,  w_rb, p_agAB, w_ab, b_root_b, out_b,  strips);
    // outp_a = xp_a@(W_root_a+W_rel_ba)^T + b_root_a   (twin branch fused)
    gemm_wmma_kernel<<<blocks, 32 * WAVES, 0, stream>>>(p_xpa, w_ca, nullptr, nullptr, b_root_a, outp_a, strips);
    // outp_b = xp_b@(W_root_b+W_rel_ab)^T + b_root_b
    gemm_wmma_kernel<<<blocks, 32 * WAVES, 0, stream>>>(p_xpb, w_cb, nullptr, nullptr, b_root_b, outp_b, strips);
}
